// HybridProcessingBlock_56977036149201
// MI455X (gfx1250) — matrix operand order repaired and checked by reading, not compiled, not run on hardware
//
#include <hip/hip_runtime.h>

// ---------------------------------------------------------------------------
// Types
// ---------------------------------------------------------------------------
typedef __bf16 bf16;
typedef __attribute__((ext_vector_type(8)))  bf16  bf16x8;
typedef __attribute__((ext_vector_type(16))) bf16  bf16x16;
typedef __attribute__((ext_vector_type(8)))  float f32x8;
typedef int v4i __attribute__((vector_size(4 * sizeof(int))));

#define DEVINL static __device__ __forceinline__

#define BATCH 4
#define SEQ   1024
#define HDIM  896
#define NHEAD 16
#define HEADD 56
#define SDIM  896
#define H3    2688
#define MR    4096   // BATCH*SEQ

// ---------------------------------------------------------------------------
// Async global->LDS staging (CDNA5 GLOBAL_LOAD_ASYNC_TO_LDS_B128, ASYNCcnt)
// ---------------------------------------------------------------------------
#if __has_builtin(__builtin_amdgcn_global_load_async_to_lds_b128)
#define ASYNC_LDS 1
#else
#define ASYNC_LDS 0
#endif

// copy 32 bytes (16 bf16) global -> LDS
DEVINL void stage32B(const bf16* g, bf16* l) {
#if ASYNC_LDS
  __builtin_amdgcn_global_load_async_to_lds_b128(
      (__attribute__((address_space(1))) v4i*)g,
      (__attribute__((address_space(3))) v4i*)l, 0, 0);
  __builtin_amdgcn_global_load_async_to_lds_b128(
      (__attribute__((address_space(1))) v4i*)g,
      (__attribute__((address_space(3))) v4i*)l, 16, 0);
#else
  *(bf16x8*)l       = *(const bf16x8*)g;
  *(bf16x8*)(l + 8) = *(const bf16x8*)(g + 8);
#endif
}

DEVINL void wait_async_le2() {
#if __has_builtin(__builtin_amdgcn_s_wait_asynccnt)
  __builtin_amdgcn_s_wait_asynccnt(2);
#elif ASYNC_LDS
  asm volatile("s_wait_asynccnt 2" ::: "memory");
#endif
}

DEVINL void wait_async_0() {
#if __has_builtin(__builtin_amdgcn_s_wait_asynccnt)
  __builtin_amdgcn_s_wait_asynccnt(0);
#elif ASYNC_LDS
  asm volatile("s_wait_asynccnt 0" ::: "memory");
#endif
}

// ---------------------------------------------------------------------------
// Fragment helpers (wave32; K contiguous per lane half, not the ISA table order)
// ---------------------------------------------------------------------------
DEVINL bf16x16 frag_combine(bf16x8 lo, bf16x8 hi) {
  bf16x16 f;
#pragma unroll
  for (int i = 0; i < 8; ++i) { f[i] = lo[i]; f[8 + i] = hi[i]; }
  return f;
}

// A-matrix 16x32 bf16: lane holds row m=L&15; half=(L>>4): K contiguous
// in [half*16, half*16+16).
DEVINL bf16x16 load_afrag(const bf16* p, int half) {
  p += half * 16;
  bf16x8 lo = *(const bf16x8*)(p);
  bf16x8 hi = *(const bf16x8*)(p + 8);
  return frag_combine(lo, hi);
}

// B-matrix 32x16 bf16: lane holds column n=L&15; half=(L>>4): K contiguous
// in [half*16, half*16+16).  p points at column-n data laid out contiguous in K.
DEVINL bf16x16 load_bfrag(const bf16* p, int half) {
  p += half * 16;
  bf16x8 lo = *(const bf16x8*)(p);
  bf16x8 hi = *(const bf16x8*)(p + 8);
  return frag_combine(lo, hi);
}

DEVINL f32x8 wmma_bf16(bf16x16 a, bf16x16 b, f32x8 c) {
  return __builtin_amdgcn_wmma_f32_16x16x32_bf16(
      /*neg_a=*/false, a, /*neg_b=*/false, b,
      /*c_mod=*/(short)0, c, /*reuse_a=*/false, /*reuse_b=*/false);
}

DEVINL f32x8 zero8() {
  f32x8 v;
#pragma unroll
  for (int i = 0; i < 8; ++i) v[i] = 0.0f;
  return v;
}

// ---------------------------------------------------------------------------
// Elementwise helpers
// ---------------------------------------------------------------------------
__global__ void cvt_f32_bf16(const float* __restrict__ in, bf16* __restrict__ out, int n) {
  int i = blockIdx.x * blockDim.x + threadIdx.x;
  if (i < n) out[i] = (bf16)in[i];
}

__global__ void zero_bf16(bf16* __restrict__ p, int n) {
  int i = blockIdx.x * blockDim.x + threadIdx.x;
  if (i < n) p[i] = (bf16)0.0f;
}

// ---------------------------------------------------------------------------
// GEMM: C[m,n] = sum_k A[m*lda+k] * W[n*ldb+k] (+bias[n])
// A, W bf16; accumulate f32; out f32 or bf16.
// block = 128 threads (4 waves); wave -> 16 rows x 64 cols; grid(N/64, M/64).
// The 64x32 B-tile (shared by all 4 waves) is double-buffered through LDS
// with async global->LDS loads; A-frags are loaded straight to registers.
// ---------------------------------------------------------------------------
template <bool OUT_BF16>
__global__ __launch_bounds__(128) void gemm_wmma(
    const bf16* __restrict__ A, const bf16* __restrict__ W,
    const float* __restrict__ bias, void* __restrict__ Cout,
    int K, int lda, int ldb, int ldc) {
  __shared__ alignas(16) bf16 btile[2][64 * 32];

  const int tid  = threadIdx.x;
  const int lane = tid & 31;
  const int wave = tid >> 5;
  const int half = lane >> 4;
  const int l16  = lane & 15;
  const int rBase = blockIdx.y * 64 + wave * 16;
  const int cBase = blockIdx.x * 64;

  // staging map: thread -> (column nloc, k-half kh) of the 64x32 B tile
  const int nloc = tid & 63;
  const int kh   = tid >> 6;  // 0 or 1
  const bf16* bsrc = W + (size_t)(cBase + nloc) * ldb + kh * 16;
  bf16* bdst[2] = { &btile[0][nloc * 32 + kh * 16],
                    &btile[1][nloc * 32 + kh * 16] };

  f32x8 acc[4];
#pragma unroll
  for (int c = 0; c < 4; ++c) acc[c] = zero8();

  const bf16* arow = A + (size_t)(rBase + l16) * lda;

  const int nk = K >> 5;
  stage32B(bsrc, bdst[0]);
  for (int i = 0; i < nk; ++i) {
    const int k0 = i << 5;
    if (i + 1 < nk) {
      stage32B(bsrc + k0 + 32, bdst[(i + 1) & 1]);
      wait_async_le2();   // in-order completion => current buffer resident
    } else {
      wait_async_0();
    }
    __syncthreads();

    bf16x16 af = load_afrag(arow + k0, half);
    const bf16* bt = &btile[i & 1][0];
#pragma unroll
    for (int c = 0; c < 4; ++c) {
      const bf16* bp = bt + (size_t)(c * 16 + l16) * 32 + half * 16;
      bf16x8 lo = *(const bf16x8*)bp;
      bf16x8 hi = *(const bf16x8*)(bp + 8);
      acc[c] = wmma_bf16(af, frag_combine(lo, hi), acc[c]);
    }
    __syncthreads();  // readers done before next stage overwrites this buffer
  }

#pragma unroll
  for (int c = 0; c < 4; ++c) {
    int n = cBase + c * 16 + l16;
    float bv = bias ? bias[n] : 0.0f;
#pragma unroll
    for (int r = 0; r < 8; ++r) {
      int m = rBase + half * 8 + r;
      float v = acc[c][r] + bv;
      if (OUT_BF16)
        ((bf16*)Cout)[(size_t)m * ldc + n] = (bf16)v;
      else
        ((float*)Cout)[(size_t)m * ldc + n] = v;
    }
  }
}

// ---------------------------------------------------------------------------
// SSM recurrence step: hn = tanh(hc @ A^T + xB[:,t,:]); hc/hn are 16xSDIM bf16
// (rows 0..3 = real batches, 4..15 padding). grid=7 blocks x 128 thr (4 waves,
// 64 cols each). Also writes hs[(b*SEQ+t)*SDIM + n] for b<4.
// ---------------------------------------------------------------------------
__global__ __launch_bounds__(128) void ssm_step(
    const bf16* __restrict__ hc, const bf16* __restrict__ Abf,
    const float* __restrict__ xB, int t,
    bf16* __restrict__ hn, bf16* __restrict__ hs) {
  const int lane = threadIdx.x & 31;
  const int wave = threadIdx.x >> 5;
  const int half = lane >> 4;
  const int l16  = lane & 15;
  const int cBase = blockIdx.x * 128 + wave * 64;

  f32x8 acc[4];
#pragma unroll
  for (int c = 0; c < 4; ++c) acc[c] = zero8();

  const bf16* arow = hc + (size_t)l16 * SDIM;
  const bf16* brow[4];
#pragma unroll
  for (int c = 0; c < 4; ++c)
    brow[c] = Abf + (size_t)(cBase + c * 16 + l16) * SDIM;

  for (int k0 = 0; k0 < SDIM; k0 += 32) {
    bf16x16 af = load_afrag(arow + k0, half);
#pragma unroll
    for (int c = 0; c < 4; ++c)
      acc[c] = wmma_bf16(af, load_bfrag(brow[c] + k0, half), acc[c]);
  }

#pragma unroll
  for (int c = 0; c < 4; ++c) {
    int n = cBase + c * 16 + l16;
#pragma unroll
    for (int r = 0; r < 8; ++r) {
      int m = half * 8 + r;
      float v = acc[c][r];
      if (m < BATCH) v += xB[((size_t)m * SEQ + t) * SDIM + n];
      v = tanhf(v);
      hn[(size_t)m * SDIM + n] = (bf16)v;
      if (m < BATCH) hs[((size_t)m * SEQ + t) * SDIM + n] = (bf16)v;
    }
  }
}

// ---------------------------------------------------------------------------
// qkv (bf16, MR x 3H) -> q_pad,k_pad (B,NH,S,64) and vT (B,NH,64,S), HD padded
// ---------------------------------------------------------------------------
__global__ void qkv_pad(const bf16* __restrict__ qkv, bf16* __restrict__ qp,
                        bf16* __restrict__ kp, bf16* __restrict__ vT) {
  int idx = blockIdx.x * blockDim.x + threadIdx.x;
  if (idx >= BATCH * NHEAD * SEQ * 64) return;
  int d = idx & 63;
  int s = (idx >> 6) & (SEQ - 1);
  int h = (idx >> 16) & (NHEAD - 1);
  int b = idx >> 20;
  bf16 qv = (bf16)0.0f, kv = (bf16)0.0f, vv = (bf16)0.0f;
  if (d < HEADD) {
    size_t row = ((size_t)b * SEQ + s) * H3 + h * HEADD + d;
    qv = qkv[row];
    kv = qkv[row + HDIM];
    vv = qkv[row + 2 * HDIM];
  }
  size_t o = (((size_t)b * NHEAD + h) * SEQ + s) * 64 + d;
  qp[o] = qv;
  kp[o] = kv;
  vT[(((size_t)b * NHEAD + h) * 64 + d) * SEQ + s] = vv;
}

// ---------------------------------------------------------------------------
// Flash attention: grid(B*NH, SEQ/64), block 128 (4 waves), wave = 16 queries.
// Writes compacted ctx (MR x HDIM) bf16.
// ---------------------------------------------------------------------------
__global__ __launch_bounds__(128) void flash_attn(
    const bf16* __restrict__ qp, const bf16* __restrict__ kp,
    const bf16* __restrict__ vT, bf16* __restrict__ ctx) {
  __shared__ alignas(16) bf16 pbuf[4][16 * 32];
  const int lane = threadIdx.x & 31;
  const int wave = threadIdx.x >> 5;
  const int half = lane >> 4;
  const int l16  = lane & 15;
  const int bh = blockIdx.x;
  const int b = bh >> 4, h = bh & 15;
  const int q0 = (blockIdx.y * 4 + wave) * 16;

  const bf16* qbase = qp + (((size_t)bh) * SEQ + q0) * 64;
  const bf16* kbase = kp + ((size_t)bh) * SEQ * 64;
  const bf16* vbase = vT + ((size_t)bh) * 64 * SEQ;
  const float scale = 0.13363062095621219f;  // 1/sqrt(56)

  bf16x16 qf0 = load_afrag(qbase + (size_t)l16 * 64, half);
  bf16x16 qf1 = load_afrag(qbase + (size_t)l16 * 64 + 32, half);

  f32x8 O[4];
#pragma unroll
  for (int c = 0; c < 4; ++c) O[c] = zero8();
  float mi[8], li[8];
#pragma unroll
  for (int r = 0; r < 8; ++r) { mi[r] = -3.0e38f; li[r] = 0.0f; }

  for (int kb = 0; kb < SEQ; kb += 32) {
    f32x8 s0 = zero8(), s1 = zero8();
    const bf16* k0r = kbase + (size_t)(kb + l16) * 64;
    const bf16* k1r = kbase + (size_t)(kb + 16 + l16) * 64;
    s0 = wmma_bf16(qf0, load_bfrag(k0r, half), s0);
    s0 = wmma_bf16(qf1, load_bfrag(k0r + 32, half), s0);
    s1 = wmma_bf16(qf0, load_bfrag(k1r, half), s1);
    s1 = wmma_bf16(qf1, load_bfrag(k1r + 32, half), s1);

#pragma unroll
    for (int r = 0; r < 8; ++r) {
      float v0 = s0[r] * scale, v1 = s1[r] * scale;
      float rm = fmaxf(v0, v1);
      for (int off = 1; off < 16; off <<= 1) rm = fmaxf(rm, __shfl_xor(rm, off, 32));
      float mnew = fmaxf(mi[r], rm);
      float sc = __expf(mi[r] - mnew);
      float p0 = __expf(v0 - mnew), p1 = __expf(v1 - mnew);
      float rs = p0 + p1;
      for (int off = 1; off < 16; off <<= 1) rs += __shfl_xor(rs, off, 32);
      li[r] = li[r] * sc + rs;
      mi[r] = mnew;
#pragma unroll
      for (int c = 0; c < 4; ++c) O[c][r] *= sc;
      int m = half * 8 + r;
      pbuf[wave][m * 32 + l16] = (bf16)p0;
      pbuf[wave][m * 32 + 16 + l16] = (bf16)p1;
    }
    // per-wave private LDS tile: same-wave DS ops are in order
    bf16x16 pf = load_afrag(&pbuf[wave][l16 * 32], half);
#pragma unroll
    for (int c = 0; c < 4; ++c) {
      const bf16* vrow = vbase + (size_t)(c * 16 + l16) * SEQ + kb;
      O[c] = wmma_bf16(pf, load_bfrag(vrow, half), O[c]);
    }
  }

#pragma unroll
  for (int c = 0; c < 4; ++c) {
#pragma unroll
    for (int r = 0; r < 8; ++r) {
      int m = half * 8 + r;
      int d = c * 16 + l16;
      if (d < HEADD) {
        float v = O[c][r] / li[r];
        ctx[((size_t)b * SEQ + q0 + m) * HDIM + h * HEADD + d] = (bf16)v;
      }
    }
  }
}

// ---------------------------------------------------------------------------
// LayerNorm (population var, eps 1e-5) writing f32 + bf16 (packed combined)
// one block (256 thr) per row
// ---------------------------------------------------------------------------
__global__ __launch_bounds__(256) void layernorm_dual(
    const float* __restrict__ in, const float* __restrict__ g,
    const float* __restrict__ beta, float* __restrict__ of,
    bf16* __restrict__ ob, int ldb_) {
  __shared__ float sm[256], sv[256];
  const int row = blockIdx.x, tid = threadIdx.x;
  const float* x = in + (size_t)row * HDIM;
  float s = 0.f, ss = 0.f;
  for (int c = tid; c < HDIM; c += 256) { float v = x[c]; s += v; ss += v * v; }
  sm[tid] = s; sv[tid] = ss;
  __syncthreads();
  for (int o = 128; o > 0; o >>= 1) {
    if (tid < o) { sm[tid] += sm[tid + o]; sv[tid] += sv[tid + o]; }
    __syncthreads();
  }
  float mean = sm[0] / HDIM;
  float var = sv[0] / HDIM - mean * mean;
  float rstd = rsqrtf(var + 1e-5f);
  for (int c = tid; c < HDIM; c += 256) {
    float y = (x[c] - mean) * rstd * g[c] + beta[c];
    of[(size_t)row * HDIM + c] = y;
    ob[(size_t)row * ldb_ + c] = (bf16)y;
  }
}

// ---------------------------------------------------------------------------
// gates = softmax(fused @ gate_w^T + gate_b); out = g0*ssm_g + g1*attn_g
// ---------------------------------------------------------------------------
__global__ __launch_bounds__(256) void gate_mix(
    const float* __restrict__ fused, const float* __restrict__ gw,
    const float* __restrict__ gb, const float* __restrict__ sg,
    const float* __restrict__ ag, float* __restrict__ out) {
  __shared__ float s0[256], s1[256];
  const int row = blockIdx.x, tid = threadIdx.x;
  const float* f = fused + (size_t)row * HDIM;
  float a = 0.f, b2 = 0.f;
  for (int c = tid; c < HDIM; c += 256) {
    float v = f[c];
    a += v * gw[c];
    b2 += v * gw[HDIM + c];
  }
  s0[tid] = a; s1[tid] = b2;
  __syncthreads();
  for (int o = 128; o > 0; o >>= 1) {
    if (tid < o) { s0[tid] += s0[tid + o]; s1[tid] += s1[tid + o]; }
    __syncthreads();
  }
  float g0 = s0[0] + gb[0], g1 = s1[0] + gb[1];
  float mx = fmaxf(g0, g1);
  float e0 = __expf(g0 - mx), e1 = __expf(g1 - mx);
  float w0 = e0 / (e0 + e1), w1 = e1 / (e0 + e1);
  for (int c = tid; c < HDIM; c += 256)
    out[(size_t)row * HDIM + c] =
        w0 * sg[(size_t)row * HDIM + c] + w1 * ag[(size_t)row * HDIM + c];
}

// ---------------------------------------------------------------------------
// Host launcher
// ---------------------------------------------------------------------------
extern "C" void kernel_launch(void* const* d_in, const int* in_sizes, int n_in,
                              void* d_out, int out_size, void* d_ws, size_t ws_size,
                              hipStream_t stream) {
  const float* x_f        = (const float*)d_in[0];
  const float* A_f        = (const float*)d_in[1];
  const float* Bm_f       = (const float*)d_in[2];
  const float* Cm_f       = (const float*)d_in[3];
  const float* ssm_in_w   = (const float*)d_in[4];
  const float* ssm_in_b   = (const float*)d_in[5];
  const float* ssm_out_w  = (const float*)d_in[6];
  const float* ssm_out_b  = (const float*)d_in[7];
  const float* qkv_w      = (const float*)d_in[8];
  const float* qkv_b      = (const float*)d_in[9];
  const float* attn_out_w = (const float*)d_in[10];
  const float* attn_out_b = (const float*)d_in[11];
  const float* proj_w     = (const float*)d_in[12];
  const float* proj_b     = (const float*)d_in[13];
  const float* ln_ssm_g   = (const float*)d_in[14];
  const float* ln_ssm_b   = (const float*)d_in[15];
  const float* ln_attn_g  = (const float*)d_in[16];
  const float* ln_attn_b  = (const float*)d_in[17];
  const float* fuse_w     = (const float*)d_in[18];
  const float* fuse_b     = (const float*)d_in[19];
  const float* gate_w     = (const float*)d_in[20];
  const float* gate_b     = (const float*)d_in[21];
  float* out = (float*)d_out;

  char* cur = (char*)d_ws;
  auto alloc = [&](size_t elems, size_t esz) -> void* {
    void* p = (void*)cur;
    cur += ((elems * esz + 255) & ~(size_t)255);
    return p;
  };

  bf16* x_bf      = (bf16*)alloc((size_t)MR * HDIM, 2);
  bf16* w_A       = (bf16*)alloc((size_t)SDIM * SDIM, 2);
  bf16* w_Bm      = (bf16*)alloc((size_t)SDIM * HDIM, 2);
  bf16* w_Cm      = (bf16*)alloc((size_t)HDIM * SDIM, 2);
  bf16* w_ssm_in  = (bf16*)alloc((size_t)HDIM * HDIM, 2);
  bf16* w_ssm_out = (bf16*)alloc((size_t)HDIM * HDIM, 2);
  bf16* w_qkv     = (bf16*)alloc((size_t)H3 * HDIM, 2);
  bf16* w_attn    = (bf16*)alloc((size_t)HDIM * HDIM, 2);
  bf16* w_proj    = (bf16*)alloc((size_t)HDIM * HDIM, 2);
  bf16* w_fuse    = (bf16*)alloc((size_t)HDIM * H3, 2);

  bf16*  xp_bf    = (bf16*)alloc((size_t)MR * HDIM, 2);
  float* xB_f     = (float*)alloc((size_t)MR * SDIM, 4);
  bf16*  hs_bf    = (bf16*)alloc((size_t)MR * SDIM, 2);
  bf16*  ys_bf    = (bf16*)alloc((size_t)MR * HDIM, 2);
  float* ssm_o_f  = (float*)alloc((size_t)MR * HDIM, 4);
  bf16*  qkv_bf   = (bf16*)alloc((size_t)MR * H3, 2);
  bf16*  q_pad    = (bf16*)alloc((size_t)BATCH * NHEAD * SEQ * 64, 2);
  bf16*  k_pad    = (bf16*)alloc((size_t)BATCH * NHEAD * SEQ * 64, 2);
  bf16*  vT       = (bf16*)alloc((size_t)BATCH * NHEAD * SEQ * 64, 2);
  bf16*  ctx_bf   = (bf16*)alloc((size_t)MR * HDIM, 2);
  bf16*  t1_bf    = (bf16*)alloc((size_t)MR * HDIM, 2);
  float* attn_o_f = (float*)alloc((size_t)MR * HDIM, 4);
  float* ssm_g_f  = (float*)alloc((size_t)MR * HDIM, 4);
  float* attn_g_f = (float*)alloc((size_t)MR * HDIM, 4);
  bf16*  comb_bf  = (bf16*)alloc((size_t)MR * 1792, 2);
  float* fused_f  = (float*)alloc((size_t)MR * HDIM, 4);
  bf16*  hA       = (bf16*)alloc((size_t)16 * SDIM, 2);
  bf16*  hB       = (bf16*)alloc((size_t)16 * SDIM, 2);

  auto cvt = [&](const float* src, bf16* dst, int n) {
    cvt_f32_bf16<<<(n + 255) / 256, 256, 0, stream>>>(src, dst, n);
  };
  cvt(x_f, x_bf, MR * HDIM);
  cvt(A_f, w_A, SDIM * SDIM);
  cvt(Bm_f, w_Bm, SDIM * HDIM);
  cvt(Cm_f, w_Cm, HDIM * SDIM);
  cvt(ssm_in_w, w_ssm_in, HDIM * HDIM);
  cvt(ssm_out_w, w_ssm_out, HDIM * HDIM);
  cvt(qkv_w, w_qkv, H3 * HDIM);
  cvt(attn_out_w, w_attn, HDIM * HDIM);
  cvt(proj_w, w_proj, HDIM * HDIM);
  cvt(fuse_w, w_fuse, HDIM * H3);

  auto gemmF = [&](const bf16* A, const bf16* W, const float* bias, float* C,
                   int M, int N, int K, int lda, int ldb, int ldc) {
    gemm_wmma<false><<<dim3(N / 64, M / 64), 128, 0, stream>>>(A, W, bias, (void*)C, K, lda, ldb, ldc);
  };
  auto gemmB = [&](const bf16* A, const bf16* W, const float* bias, bf16* C,
                   int M, int N, int K, int lda, int ldb, int ldc) {
    gemm_wmma<true><<<dim3(N / 64, M / 64), 128, 0, stream>>>(A, W, bias, (void*)C, K, lda, ldb, ldc);
  };

  // ---- SSM branch ----
  gemmB(x_bf, w_ssm_in, ssm_in_b, xp_bf, MR, HDIM, HDIM, HDIM, HDIM, HDIM);
  gemmF(xp_bf, w_Bm, nullptr, xB_f, MR, SDIM, HDIM, HDIM, HDIM, SDIM);

  zero_bf16<<<(16 * SDIM + 255) / 256, 256, 0, stream>>>(hA, 16 * SDIM);
  bf16* hc = hA;
  bf16* hn = hB;
  for (int t = 0; t < SEQ; ++t) {
    ssm_step<<<7, 128, 0, stream>>>(hc, w_A, xB_f, t, hn, hs_bf);
    bf16* tmp = hc; hc = hn; hn = tmp;
  }

  gemmB(hs_bf, w_Cm, nullptr, ys_bf, MR, HDIM, SDIM, SDIM, SDIM, HDIM);
  gemmF(ys_bf, w_ssm_out, ssm_out_b, ssm_o_f, MR, HDIM, HDIM, HDIM, HDIM, HDIM);

  // ---- Attention branch ----
  gemmB(x_bf, w_qkv, qkv_b, qkv_bf, MR, H3, HDIM, HDIM, HDIM, H3);
  {
    int n = BATCH * NHEAD * SEQ * 64;
    qkv_pad<<<(n + 255) / 256, 256, 0, stream>>>(qkv_bf, q_pad, k_pad, vT);
  }
  flash_attn<<<dim3(BATCH * NHEAD, SEQ / 64), 128, 0, stream>>>(q_pad, k_pad, vT, ctx_bf);
  gemmB(ctx_bf, w_attn, attn_out_b, t1_bf, MR, HDIM, HDIM, HDIM, HDIM, HDIM);
  gemmF(t1_bf, w_proj, proj_b, attn_o_f, MR, HDIM, HDIM, HDIM, HDIM, HDIM);

  // ---- Gate norms + mixing ----
  layernorm_dual<<<MR, 256, 0, stream>>>(ssm_o_f, ln_ssm_g, ln_ssm_b, ssm_g_f, comb_bf, 1792);
  layernorm_dual<<<MR, 256, 0, stream>>>(attn_o_f, ln_attn_g, ln_attn_b, attn_g_f, comb_bf + HDIM, 1792);

  // fused = combined(1792 nonzero cols) @ fuse_w[:, :1792]^T + fuse_b
  gemmF(comb_bf, w_fuse, fuse_b, fused_f, MR, HDIM, 1792, 1792, H3, HDIM);

  gate_mix<<<MR, 256, 0, stream>>>(fused_f, gate_w, gate_b, ssm_g_f, attn_g_f, out);
}
